// SelfAttention_55765855371452
// MI455X (gfx1250) — matrix operand order repaired and checked by reading, not compiled, not run on hardware
//
#include <hip/hip_runtime.h>
#include <hip/hip_bf16.h>

// ---------------- problem constants ----------------
#define B_   2
#define T_   2048
#define C_   1024
#define H_   16
#define DH_  64
#define M_   (B_ * T_)     // 4096 rows of x
#define N3_  (3 * C_)      // 3072 qkv cols

typedef __attribute__((ext_vector_type(16))) _Float16 v16h;
typedef __attribute__((ext_vector_type(8)))  float    v8f;
typedef __attribute__((ext_vector_type(4)))  int      v4i_t;

// ---------------- async global -> LDS staging (CDNA5 ASYNCcnt path) --------
#if defined(__has_builtin)
#if __has_builtin(__builtin_amdgcn_global_load_async_to_lds_b128) && \
    __has_builtin(__builtin_amdgcn_s_wait_asynccnt)
#define HAVE_ASYNC_LDS 1
#endif
#endif
#ifndef HAVE_ASYNC_LDS
#define HAVE_ASYNC_LDS 0
#endif

typedef __attribute__((address_space(1))) v4i_t* gb128_ptr;   // global b128
typedef __attribute__((address_space(3))) v4i_t* lb128_ptr;   // LDS b128

static __device__ __forceinline__ void async_copy_b128(_Float16* lds_dst,
                                                       const _Float16* gsrc) {
#if HAVE_ASYNC_LDS
  __builtin_amdgcn_global_load_async_to_lds_b128(
      (gb128_ptr)(void*)gsrc, (lb128_ptr)(void*)lds_dst, 0, 0);
#else
  *(v4i_t*)lds_dst = *(const v4i_t*)gsrc;
#endif
}
static __device__ __forceinline__ void async_wait() {
#if HAVE_ASYNC_LDS
  __builtin_amdgcn_s_wait_asynccnt(0);
#endif
}

// ---------------- WMMA wrapper ----------------
static __device__ __forceinline__ v8f wmma_f16(v16h a, v16h b, v8f c) {
  // (neg_a, A, neg_b, B, c_mod, C, reuse_a, reuse_b)
  return __builtin_amdgcn_wmma_f32_16x16x32_f16(false, a, false, b, (short)0, c,
                                                false, false);
}

// A fragment (16x32 f16, row-major source, leading dim ld):
// lane l holds row m=l&15; VGPR v holds K = (v>=4?16:0) + (l>=16?8:0) + (v&3)*2 (+0,+1)
static __device__ __forceinline__ v16h load_frag_a(const _Float16* base, int ld) {
  const int lane = threadIdx.x & 31;
  const _Float16* row = base + (lane & 15) * ld;
  const int kh = (lane >> 4) << 3;  // 0 or 8
  v16h f;
#pragma unroll
  for (int v = 0; v < 8; ++v) {
    const int k = ((v >> 2) << 4) + kh + ((v & 3) << 1);
    f[2 * v]     = row[k];
    f[2 * v + 1] = row[k + 1];
  }
  return f;
}

// B fragment (32x16 f16). Source stored "transposed": row n of baseT = col n of B.
// lane l -> col n=l&15; VGPR v holds K = (v>=4?16:0) + (l>=16?8:0) + (v&3)*2 (+0,+1)
static __device__ __forceinline__ v16h load_frag_b(const _Float16* baseT, int ld) {
  const int lane = threadIdx.x & 31;
  const _Float16* row = baseT + (lane & 15) * ld;
  const int kh = (lane >> 4) << 3;  // 0 or 8
  v16h f;
#pragma unroll
  for (int v = 0; v < 8; ++v) {
    const int k = ((v >> 2) << 4) + kh + ((v & 3) << 1);
    f[2 * v]     = row[k];
    f[2 * v + 1] = row[k + 1];
  }
  return f;
}

// wave-internal LDS producer->consumer ordering (P re-layout tile is wave-private)
static __device__ __forceinline__ void wave_lds_sync() {
  __builtin_amdgcn_wave_barrier();
#if defined(__has_builtin)
#if __has_builtin(__builtin_amdgcn_s_wait_dscnt)
  __builtin_amdgcn_s_wait_dscnt(0);
#endif
#endif
  __builtin_amdgcn_wave_barrier();
}

// 16-lane-half reductions (wave32: one C-fragment row = one VGPR across 16 lanes)
static __device__ __forceinline__ float hmax16(float v) {
  v = fmaxf(v, __shfl_xor(v, 8, 32));
  v = fmaxf(v, __shfl_xor(v, 4, 32));
  v = fmaxf(v, __shfl_xor(v, 2, 32));
  v = fmaxf(v, __shfl_xor(v, 1, 32));
  return v;
}
static __device__ __forceinline__ float hsum16(float v) {
  v += __shfl_xor(v, 8, 32);
  v += __shfl_xor(v, 4, 32);
  v += __shfl_xor(v, 2, 32);
  v += __shfl_xor(v, 1, 32);
  return v;
}

// ---------------- prep kernels ----------------
__global__ void cvt_f32_to_f16(const float* __restrict__ in,
                               _Float16* __restrict__ out, int n) {
  int i = blockIdx.x * blockDim.x + threadIdx.x;
  if (i < n) out[i] = (_Float16)in[i];
}

// out[n*K + k] = (f16) in[k*N + n]   (store W transposed, N-major)
__global__ void transpose_to_f16(const float* __restrict__ in,
                                 _Float16* __restrict__ out, int K, int N) {
  int i = blockIdx.x * blockDim.x + threadIdx.x;
  if (i < K * N) {
    int n = i / K, k = i - n * K;
    out[i] = (_Float16)in[(size_t)k * N + n];
  }
}

// ---------------- QKV projection GEMM ----------------
// block (4 waves) computes 64x64 tile; A/B k-slices async-staged in LDS.
// scatter: cols [0,1024) -> Q(b,h,t,d); [1024,2048) -> K(b,h,t,d);
//          [2048,3072) -> Vt(b,h,d,t)  (transposed for the P*V B-fragments)
__global__ __launch_bounds__(128) void qkv_gemm(
    const _Float16* __restrict__ xh, const _Float16* __restrict__ wt,
    const float* __restrict__ bias, _Float16* __restrict__ Qb,
    _Float16* __restrict__ Kb, _Float16* __restrict__ Vt) {
  __shared__ __align__(16) _Float16 atile[64 * 40];
  __shared__ __align__(16) _Float16 btile[64 * 40];

  const int tid = threadIdx.x;
  const int lane = tid & 31;
  const int w = tid >> 5;
  const int NT = N3_ / 64;                 // 48
  const int mt = blockIdx.x / NT, nt = blockIdx.x - mt * NT;
  const int m0 = mt * 64, n0 = nt * 64;

  const int srow = tid >> 1;               // 0..63
  const int spart = (tid & 1) << 4;        // 0 or 16 halves

  v8f zero = {};
  v8f acc[4] = {zero, zero, zero, zero};
  for (int k0 = 0; k0 < C_; k0 += 32) {
    async_copy_b128(atile + srow * 40 + spart,
                    xh + (size_t)(m0 + srow) * C_ + k0 + spart);
    async_copy_b128(btile + srow * 40 + spart,
                    wt + (size_t)(n0 + srow) * C_ + k0 + spart);
    async_wait();
    __syncthreads();
    v16h a = load_frag_a(atile + 16 * w * 40, 40);
#pragma unroll
    for (int f = 0; f < 4; ++f) {
      v16h b = load_frag_b(btile + 16 * f * 40, 40);
      acc[f] = wmma_f16(a, b, acc[f]);
    }
    __syncthreads();
  }

  const int nl = lane & 15;
  const int mhi = (lane >> 4) << 3;
  const int region = n0 >> 10;             // uniform per block
#pragma unroll
  for (int f = 0; f < 4; ++f) {
    const int n  = n0 + 16 * f + nl;
    const float bv = bias[n];
    const int nn = n & (C_ - 1);
    const int h = nn >> 6, d = nn & 63;
#pragma unroll
    for (int v = 0; v < 8; ++v) {
      const int row = m0 + 16 * w + v + mhi;   // 0..4095
      const int bb = row >> 11;                // /T_
      const int t  = row & (T_ - 1);
      const int bh = bb * H_ + h;
      const _Float16 val = (_Float16)(acc[f][v] + bv);
      if (region == 0)
        Qb[((size_t)bh * T_ + t) * DH_ + d] = val;
      else if (region == 1)
        Kb[((size_t)bh * T_ + t) * DH_ + d] = val;
      else
        Vt[((size_t)bh * DH_ + d) * T_ + t] = val;
    }
  }
}

// ---------------- flash attention ----------------
// block (4 waves) owns 64 consecutive query rows of one (b,h); K/V tiles for
// each 32-key step are async-staged in LDS and shared by all 4 waves.
// Trip count is block-uniform; fully-masked steps contribute exactly zero.
__global__ __launch_bounds__(128) void attn_flash(
    const _Float16* __restrict__ Qb, const _Float16* __restrict__ Kb,
    const _Float16* __restrict__ Vt, _Float16* __restrict__ ah) {
  __shared__ __align__(16) _Float16 ktile[32 * 72];      // 32 keys x 64 d
  __shared__ __align__(16) _Float16 vtile[64 * 40];      // 64 d x 32 keys
  __shared__ __align__(16) _Float16 ptile[4][16 * 40];   // per-wave P re-layout

  const int tid = threadIdx.x;
  const int lane = tid & 31;
  const int w    = tid >> 5;
  const int qb   = blockIdx.x & (T_ / 64 - 1);  // 0..31
  const int bh   = blockIdx.x >> 5;             // 0..31
  const int q0   = qb * 64 + w * 16;

  const _Float16* Qp = Qb + (size_t)bh * T_ * DH_;
  const _Float16* Kp = Kb + (size_t)bh * T_ * DH_;
  const _Float16* Vp = Vt + (size_t)bh * DH_ * T_;
  _Float16* pt = ptile[w];

  const v16h qa0 = load_frag_a(Qp + (size_t)q0 * DH_ + 0, DH_);
  const v16h qa1 = load_frag_a(Qp + (size_t)q0 * DH_ + 32, DH_);

  v8f zero = {};
  v8f o[4] = {zero, zero, zero, zero};
  float rmax[8], rsum[8];
#pragma unroll
  for (int v = 0; v < 8; ++v) { rmax[v] = -1.0e30f; rsum[v] = 0.0f; }

  const int nl = lane & 15;
  const int mhi = (lane >> 4) << 3;
  const float scale = 0.125f;                  // 1/sqrt(DH)
  const int jmax = qb * 2 + 1;                 // covers rows up to qb*64+63

  const int krow = tid >> 2, kpart = (tid & 3) << 4;   // K stage: 32 rows x 4 chunks
  const int vrow = tid >> 1, vpart = (tid & 1) << 4;   // V stage: 64 rows x 2 chunks

  for (int j = 0; j <= jmax; ++j) {
    const int s0 = j * 32;
    async_copy_b128(ktile + krow * 72 + kpart,
                    Kp + (size_t)(s0 + krow) * DH_ + kpart);
    async_copy_b128(vtile + vrow * 40 + vpart,
                    Vp + (size_t)vrow * T_ + s0 + vpart);
    async_wait();
    __syncthreads();

    // ---- S = Q * K^T  (16 x 32, two C fragments), K read from LDS ----
    v8f sf0 = zero, sf1 = zero;
    sf0 = wmma_f16(qa0, load_frag_b(ktile, 72), sf0);
    sf0 = wmma_f16(qa1, load_frag_b(ktile + 32, 72), sf0);
    sf1 = wmma_f16(qa0, load_frag_b(ktile + 16 * 72, 72), sf1);
    sf1 = wmma_f16(qa1, load_frag_b(ktile + 16 * 72 + 32, 72), sf1);

    // ---- causal mask + online softmax (fp32) ----
#pragma unroll
    for (int v = 0; v < 8; ++v) {
      const int row = q0 + v + mhi;
      float a = sf0[v] * scale;
      float b = sf1[v] * scale;
      if (s0 + nl > row)      a = -1.0e30f;
      if (s0 + 16 + nl > row) b = -1.0e30f;
      float mx = hmax16(fmaxf(a, b));
      const float mnew = fmaxf(rmax[v], mx);
      const float esc  = __expf(rmax[v] - mnew);
      const float p0   = __expf(a - mnew);
      const float p1   = __expf(b - mnew);
      rsum[v] = rsum[v] * esc + hsum16(p0 + p1);
      rmax[v] = mnew;
      o[0][v] *= esc; o[1][v] *= esc; o[2][v] *= esc; o[3][v] *= esc;
      pt[(v + mhi) * 40 + nl]      = (_Float16)p0;
      pt[(v + mhi) * 40 + nl + 16] = (_Float16)p1;
    }
    wave_lds_sync();
    const v16h pa = load_frag_a(pt, 40);       // C-layout -> A-layout via LDS

    // ---- O += P * V  (V read from LDS, stored d-major) ----
#pragma unroll
    for (int f = 0; f < 4; ++f) {
      v16h vb = load_frag_b(vtile + 16 * f * 40, 40);
      o[f] = wmma_f16(pa, vb, o[f]);
    }
    __syncthreads();
  }

  // ---- normalize + write attention output (b, t, h*64+d) as f16 ----
  const int b = bh >> 4, h = bh & (H_ - 1);
#pragma unroll
  for (int v = 0; v < 8; ++v) {
    const float inv = 1.0f / rsum[v];
    const int t = q0 + v + mhi;
    _Float16* dst = ah + ((size_t)b * T_ + t) * C_ + h * DH_;
#pragma unroll
    for (int f = 0; f < 4; ++f) dst[16 * f + nl] = (_Float16)(o[f][v] * inv);
  }
}

// ---------------- output projection GEMM (f16 in, fp32 out) ----------------
__global__ __launch_bounds__(128) void out_gemm(
    const _Float16* __restrict__ ah, const _Float16* __restrict__ wt,
    const float* __restrict__ bias, float* __restrict__ out) {
  __shared__ __align__(16) _Float16 atile[64 * 40];
  __shared__ __align__(16) _Float16 btile[64 * 40];

  const int tid = threadIdx.x;
  const int lane = tid & 31;
  const int w = tid >> 5;
  const int NT = C_ / 64;                  // 16
  const int mt = blockIdx.x / NT, nt = blockIdx.x - mt * NT;
  const int m0 = mt * 64, n0 = nt * 64;

  const int srow = tid >> 1;
  const int spart = (tid & 1) << 4;

  v8f zero = {};
  v8f acc[4] = {zero, zero, zero, zero};
  for (int k0 = 0; k0 < C_; k0 += 32) {
    async_copy_b128(atile + srow * 40 + spart,
                    ah + (size_t)(m0 + srow) * C_ + k0 + spart);
    async_copy_b128(btile + srow * 40 + spart,
                    wt + (size_t)(n0 + srow) * C_ + k0 + spart);
    async_wait();
    __syncthreads();
    v16h a = load_frag_a(atile + 16 * w * 40, 40);
#pragma unroll
    for (int f = 0; f < 4; ++f) {
      v16h b = load_frag_b(btile + 16 * f * 40, 40);
      acc[f] = wmma_f16(a, b, acc[f]);
    }
    __syncthreads();
  }

  const int nl = lane & 15;
  const int mhi = (lane >> 4) << 3;
#pragma unroll
  for (int f = 0; f < 4; ++f) {
    const int n = n0 + 16 * f + nl;
    const float bv = bias[n];
#pragma unroll
    for (int v = 0; v < 8; ++v) {
      const int row = m0 + 16 * w + v + mhi;
      out[(size_t)row * C_ + n] = acc[f][v] + bv;
    }
  }
}

// ---------------- host side ----------------
extern "C" void kernel_launch(void* const* d_in, const int* in_sizes, int n_in,
                              void* d_out, int out_size, void* d_ws, size_t ws_size,
                              hipStream_t stream) {
  const float* x     = (const float*)d_in[0];
  const float* Wqkv  = (const float*)d_in[1];
  const float* bqkv  = (const float*)d_in[2];
  const float* Wout  = (const float*)d_in[3];
  const float* bout  = (const float*)d_in[4];
  float* out = (float*)d_out;

  // workspace layout (f16 halves), total ~48 MB
  _Float16* ws    = (_Float16*)d_ws;
  _Float16* xh    = ws;                                   // M_*C_
  _Float16* wqkvt = xh    + (size_t)M_ * C_;              // N3_*C_
  _Float16* wot   = wqkvt + (size_t)N3_ * C_;             // C_*C_
  _Float16* Qb    = wot   + (size_t)C_ * C_;              // B*H*T*DH
  _Float16* Kb    = Qb    + (size_t)B_ * H_ * T_ * DH_;
  _Float16* Vt    = Kb    + (size_t)B_ * H_ * T_ * DH_;
  _Float16* ah    = Vt    + (size_t)B_ * H_ * T_ * DH_;   // M_*C_

  // 1) precision conversion / weight transposition
  {
    int n = M_ * C_;
    cvt_f32_to_f16<<<n / 256, 256, 0, stream>>>(x, xh, n);
  }
  {
    int n = C_ * N3_;
    transpose_to_f16<<<n / 256, 256, 0, stream>>>(Wqkv, wqkvt, C_, N3_);
  }
  {
    int n = C_ * C_;
    transpose_to_f16<<<n / 256, 256, 0, stream>>>(Wout, wot, C_, C_);
  }

  // 2) QKV projection: (M_/64)*(N3_/64) = 3072 blocks of 4 waves
  qkv_gemm<<<(M_ / 64) * (N3_ / 64), 128, 0, stream>>>(xh, wqkvt, bqkv,
                                                       Qb, Kb, Vt);

  // 3) flash attention: B*H*(T/64) = 1024 blocks of 4 waves
  attn_flash<<<B_ * H_ * (T_ / 64), 128, 0, stream>>>(Qb, Kb, Vt, ah);

  // 4) output projection: (M_/64)*(C_/64) = 1024 blocks of 4 waves
  out_gemm<<<(M_ / 64) * (C_ / 64), 128, 0, stream>>>(ah, wot, bout, out);
}
